// CausalSelfMHAttention_34162169872543
// MI455X (gfx1250) — matrix operand order repaired and checked by reading, not compiled, not run on hardware
//
#include <hip/hip_runtime.h>

#define T_SEQ 8192
#define CE    1024
#define NHEAD 16

typedef __attribute__((ext_vector_type(16))) __bf16       v16bf;
typedef __attribute__((ext_vector_type(8)))  float        v8f;
typedef __attribute__((ext_vector_type(4)))  unsigned int v4u;
typedef __attribute__((ext_vector_type(4)))  unsigned int u32x4;
typedef __attribute__((ext_vector_type(8)))  int          i32x8;
typedef __attribute__((ext_vector_type(4)))  int          i32x4;

union Frag { v16bf v; v4u q[2]; };

__device__ __forceinline__ unsigned short f2bf(float f) {
  unsigned int u = __float_as_uint(f);
  u += 0x7FFFu + ((u >> 16) & 1u);
  return (unsigned short)(u >> 16);
}
__device__ __forceinline__ float bf2f(unsigned short h) {
  return __uint_as_float(((unsigned int)h) << 16);
}
__device__ __forceinline__ v8f v8f_zero() {
  v8f z;
#pragma unroll
  for (int i = 0; i < 8; i++) z[i] = 0.0f;
  return z;
}
#define WMMA_BF16(A, B, C) \
  __builtin_amdgcn_wmma_f32_16x16x32_bf16(false, (A), false, (B), (short)0, (C), false, false)

// ---------------------------------------------------------------------------
// fp32 -> bf16 (RNE) conversion
// ---------------------------------------------------------------------------
__global__ void cvt_f32_bf16(const float* __restrict__ in,
                             unsigned short* __restrict__ out, int n) {
  int i = blockIdx.x * blockDim.x + threadIdx.x;
  if (i < n) out[i] = f2bf(in[i]);
}

// ---------------------------------------------------------------------------
// bf16 WMMA GEMM: C[M,N] = A[M,K] * B[K,N] + bias[N]
// block tile 128x128, K-step 32; 8 waves (2x4), each wave 64x32 (4x2 tiles).
// A tile is DMA'd into LDS by the Tensor Data Mover (wave 0 issues the
// descriptor, waits on TENSORcnt, workgroup barrier publishes it).
// B tile is loaded by all threads and stored transposed ([n][k]) so each
// lane's 16-element B fragment read is contiguous in LDS.
// ---------------------------------------------------------------------------
__global__ __launch_bounds__(256) void gemm_bf16_wmma(
    const unsigned short* __restrict__ A, const unsigned short* __restrict__ B,
    const float* __restrict__ bias, void* __restrict__ C,
    int M, int N, int K, int c_f32) {
  __shared__ __align__(16) unsigned short As[128 * 32];  // [m][k] packed (TDM image)
  __shared__ __align__(16) unsigned short Bs[128 * 40];  // [n][k], stride 40

  const int tid = threadIdx.x;
  const int lane = tid & 31;
  const int l16 = lane & 15;
  const int hi = (lane < 16) ? 0 : 1;
  const int wave = tid >> 5;
  const int wm = wave >> 2;  // 0..1
  const int wn = wave & 3;   // 0..3
  const int blockRow = blockIdx.y * 128;
  const int blockCol = blockIdx.x * 128;
  const unsigned ldsA = (unsigned)(unsigned long long)(const void*)As;

  v8f acc[4][2];
#pragma unroll
  for (int i = 0; i < 4; i++)
#pragma unroll
    for (int j = 0; j < 2; j++) acc[i][j] = v8f_zero();

  for (int kt = 0; kt < K; kt += 32) {
    if (wave == 0) {
      // ---- Tensor Data Mover: 128x32 bf16 tile of A -> LDS (packed) ----
      unsigned long long ga =
          (unsigned long long)(const void*)(A + (size_t)blockRow * K + kt);
      u32x4 g0;
      g0[0] = 1u;                                   // count=1, user descriptor
      g0[1] = ldsA;                                 // LDS byte address
      g0[2] = (unsigned)ga;                         // global addr [31:0]
      g0[3] = (unsigned)((ga >> 32) & 0x01FFFFFFull) | 0x80000000u;  // [56:32] | type=2
      i32x8 g1;
      g1[0] = 0x00010000;                                        // data_size = 2B
      g1[1] = (int)(((unsigned)K & 0xFFFFu) << 16);              // tensor_dim0 lo
      g1[2] = (int)((((unsigned)K) >> 16) | (((unsigned)M & 0xFFFFu) << 16));
      g1[3] = (int)((((unsigned)M) >> 16) | (32u << 16));        // tile_dim0 = 32
      g1[4] = 128;                                               // tile_dim1 = 128
      g1[5] = K;                                                 // dim0 stride lo
      g1[6] = 0;
      g1[7] = 0;
      i32x4 gz4;
      gz4[0] = 0; gz4[1] = 0; gz4[2] = 0; gz4[3] = 0;            // 2-D tensor
      i32x8 gz8;
#pragma unroll
      for (int z = 0; z < 8; z++) gz8[z] = 0;
      __builtin_amdgcn_tensor_load_to_lds(g0, g1, gz4, gz4, gz8, 0);
    }
    {  // B tile 32x128 -> transposed Bs[n][k]
      int k = tid >> 3;
      int nseg = (tid & 7) * 16;
      const unsigned short* gp = B + (size_t)(kt + k) * N + blockCol + nseg;
      __builtin_prefetch(gp + (size_t)32 * N, 0, 1);
      v4u d0 = *(const v4u*)(gp);
      v4u d1 = *(const v4u*)(gp + 8);
      const unsigned short* s0 = (const unsigned short*)&d0;
      const unsigned short* s1 = (const unsigned short*)&d1;
#pragma unroll
      for (int i = 0; i < 8; i++) Bs[(nseg + i) * 40 + k] = s0[i];
#pragma unroll
      for (int i = 0; i < 8; i++) Bs[(nseg + 8 + i) * 40 + k] = s1[i];
    }
    if (wave == 0) __builtin_amdgcn_s_wait_tensorcnt(0);
    __syncthreads();

    Frag afr[4], bfr[2];
    const int ka0 = hi * 16;  // A: lanes>=16 hold K 16..31
    const int kb0 = hi * 16;  // B: lanes>=16 hold K 16..31
#pragma unroll
    for (int mt = 0; mt < 4; mt++) {
      int row = wm * 64 + mt * 16 + l16;
      afr[mt].q[0] = *(const v4u*)&As[row * 32 + ka0];
      afr[mt].q[1] = *(const v4u*)&As[row * 32 + ka0 + 8];
    }
#pragma unroll
    for (int nt = 0; nt < 2; nt++) {
      int col = wn * 32 + nt * 16 + l16;
      bfr[nt].q[0] = *(const v4u*)&Bs[col * 40 + kb0];
      bfr[nt].q[1] = *(const v4u*)&Bs[col * 40 + kb0 + 8];
    }
#pragma unroll
    for (int mt = 0; mt < 4; mt++)
#pragma unroll
      for (int nt = 0; nt < 2; nt++)
        acc[mt][nt] = WMMA_BF16(afr[mt].v, bfr[nt].v, acc[mt][nt]);
    __syncthreads();
  }

  // ---- epilogue: one uniform branch, bias loaded once per column tile ----
  if (c_f32) {
    float* Co = (float*)C;
#pragma unroll
    for (int mt = 0; mt < 4; mt++)
#pragma unroll
      for (int nt = 0; nt < 2; nt++) {
        int col = blockCol + wn * 32 + nt * 16 + l16;
        float bv = bias[col];
#pragma unroll
        for (int r = 0; r < 8; r++) {
          int row = blockRow + wm * 64 + mt * 16 + r + hi * 8;
          Co[(size_t)row * N + col] = acc[mt][nt][r] + bv;
        }
      }
  } else {
    unsigned short* Co = (unsigned short*)C;
#pragma unroll
    for (int mt = 0; mt < 4; mt++)
#pragma unroll
      for (int nt = 0; nt < 2; nt++) {
        int col = blockCol + wn * 32 + nt * 16 + l16;
        float bv = bias[col];
#pragma unroll
        for (int r = 0; r < 8; r++) {
          int row = blockRow + wm * 64 + mt * 16 + r + hi * 8;
          Co[(size_t)row * N + col] = f2bf(acc[mt][nt][r] + bv);
        }
      }
  }
}

// ---------------------------------------------------------------------------
// Pooled K/V means. grid = (2016, 64 heads), block = 64 (one lane per d).
// Pooled row layout per head: levels 1..6 concatenated (1024,512,256,128,64,32)
// ---------------------------------------------------------------------------
__global__ void pool_kv(const unsigned short* __restrict__ qkv,  // [B*T, 3C] bf16
                        unsigned short* __restrict__ pk,
                        unsigned short* __restrict__ pv) {
  const int off[7] = {0, 1024, 1536, 1792, 1920, 1984, 2016};
  int rowFlat = blockIdx.x;  // 0..2015
  int bh = blockIdx.y;       // 0..63
  int d = threadIdx.x;       // 0..63
  int l = 0;
  while (l < 5 && rowFlat >= off[l + 1]) l++;  // level-1 in 0..5
  int c = rowFlat - off[l];
  int r = 8 << l;  // pool width 8..256
  int b = bh >> 4, h = bh & 15;
  size_t base = ((size_t)b * T_SEQ + (size_t)c * r) * (3 * CE) + h * 64 + d;
  float sk = 0.0f, sv = 0.0f;
  for (int j = 0; j < r; j++) {
    sk += bf2f(qkv[base + (size_t)j * (3 * CE) + CE]);
    sv += bf2f(qkv[base + (size_t)j * (3 * CE) + 2 * CE]);
  }
  float inv = 1.0f / (float)r;
  size_t o = ((size_t)bh * 2016 + rowFlat) * 64 + d;
  pk[o] = f2bf(sk * inv);
  pv[o] = f2bf(sv * inv);
}

// ---------------------------------------------------------------------------
// Fused FMA attention. grid=(128 q-blocks, 64 heads), block=128 (4 waves).
// Wave w owns query rows [16w,16w+16) and the full 224-key score stripe.
// LDS (64KB dynamic): Qs 64x64 | Ks 128x64 (->P0 64x128) | Kcs 96x64
// (->Pc 64x96) | Vst 64x128 (transposed) | Vcst 64x96 (transposed)
// ---------------------------------------------------------------------------
__global__ __launch_bounds__(128) void fma_attn(
    const unsigned short* __restrict__ qkv,
    const unsigned short* __restrict__ pk,
    const unsigned short* __restrict__ pv,
    unsigned short* __restrict__ o) {  // [B*T, C] bf16
  extern __shared__ __align__(16) unsigned short sm[];
  unsigned short* Qs = sm;             // 64*64
  unsigned short* Ks = sm + 4096;      // 128*64
  unsigned short* Kcs = sm + 12288;    // 96*64
  unsigned short* Vst = sm + 18432;    // 64*128   [d][key]
  unsigned short* Vcst = sm + 26624;   // 64*96    [d][key]
  unsigned short* P0 = Ks;             // overlay: 64*128 probs (level 0)
  unsigned short* Pc = Kcs;            // overlay: 64*96 probs (coarse)

  const int g = blockIdx.x;   // 0..127 query block
  const int bh = blockIdx.y;  // 0..63
  const int b = bh >> 4, h = bh & 15;
  const int tid = threadIdx.x;
  const int lane = tid & 31;
  const int w = tid >> 5;
  const int l16 = lane & 15;
  const int hi = (lane < 16) ? 0 : 1;
  const size_t rs = 3 * CE;  // qkv row stride
  const size_t hOff = (size_t)h * 64;
  const int off6[6] = {0, 1024, 1536, 1792, 1920, 1984};

  // ---- load Q (pre-scaled by 1/sqrt(D)=0.125) ----
#pragma unroll
  for (int i = 0; i < 32; i++) {
    int idx = tid * 32 + i;  // 4096 elems
    int row = idx >> 6, col = idx & 63;
    size_t t = (size_t)g * 64 + row;
    float qv = bf2f(qkv[((size_t)b * T_SEQ + t) * rs + hOff + col]) * 0.125f;
    Qs[row * 64 + col] = f2bf(qv);
  }
  // ---- load fine K (rows 0..63 = block g-1, 64..127 = block g) and V^T ----
#pragma unroll
  for (int i = 0; i < 64; i++) {
    int idx = tid * 64 + i;  // 8192 elems
    int rk = idx >> 6, col = idx & 63;
    int t = (rk < 64) ? ((g - 1) * 64 + rk) : (g * 64 + rk - 64);
    if (t < 0) t = 0;  // masked later
    size_t gbase = ((size_t)b * T_SEQ + t) * rs + hOff + col;
    Ks[rk * 64 + col] = qkv[gbase + CE];
    Vst[col * 128 + rk] = qkv[gbase + 2 * CE];
  }
  // ---- load coarse K, V^T: 96 rows = 6 levels x (8 keys of G-2, 8 of G-1) ---
#pragma unroll
  for (int i = 0; i < 48; i++) {
    int idx = tid * 48 + i;  // 6144 elems
    int j = idx >> 6, col = idx & 63;
    int l = (j >> 4) + 1;  // 1..6
    int n = j & 15;
    int sblk = (n < 8) ? 2 : 1;
    int p = n & 7;
    int G = g >> (l - 1);
    int c = (G - sblk) * 8 + p;
    int rows_l = 2048 >> l;  // #coarse rows at level l
    if (c < 0) c = 0;
    if (c >= rows_l) c = rows_l - 1;  // masked later
    size_t po = ((size_t)bh * 2016 + off6[l - 1] + c) * 64 + col;
    Kcs[j * 64 + col] = pk[po];
    Vcst[col * 96 + j] = pv[po];
  }
  __syncthreads();

  // ---- scores: 14 column tiles (4 prev + 4 self + 6 coarse) ----
  Frag qf[2];
  {
    int row = w * 16 + l16;
    int k0 = hi * 16;
    qf[0].q[0] = *(const v4u*)&Qs[row * 64 + k0];
    qf[0].q[1] = *(const v4u*)&Qs[row * 64 + k0 + 8];
    qf[1].q[0] = *(const v4u*)&Qs[row * 64 + 32 + k0];
    qf[1].q[1] = *(const v4u*)&Qs[row * 64 + 32 + k0 + 8];
  }
  v8f sc[14];
  const int kb0 = hi * 16;
#pragma unroll
  for (int ct = 0; ct < 14; ct++) {
    const unsigned short* kbase = (ct < 8) ? Ks : Kcs;
    int jrow = ((ct < 8) ? ct * 16 : (ct - 8) * 16) + l16;
    Frag bf0, bf1;
    bf0.q[0] = *(const v4u*)&kbase[jrow * 64 + kb0];
    bf0.q[1] = *(const v4u*)&kbase[jrow * 64 + kb0 + 8];
    bf1.q[0] = *(const v4u*)&kbase[jrow * 64 + 32 + kb0];
    bf1.q[1] = *(const v4u*)&kbase[jrow * 64 + 32 + kb0 + 8];
    v8f a = v8f_zero();
    a = WMMA_BF16(qf[0].v, bf0.v, a);
    a = WMMA_BF16(qf[1].v, bf1.v, a);
    sc[ct] = a;
  }
  // ---- masks ----
#pragma unroll
  for (int ct = 0; ct < 14; ct++) {
#pragma unroll
    for (int r = 0; r < 8; r++) {
      int i_loc = w * 16 + r + hi * 8;
      int n = l16;
      bool valid;
      if (ct < 4)
        valid = ((ct * 16 + n) > i_loc) && (g >= 1);  // prev-block window
      else if (ct < 8)
        valid = ((ct - 4) * 16 + n) <= i_loc;  // self-block causal
      else {
        int l = ct - 7;
        int sblk = (n < 8) ? 2 : 1;
        valid = (g >> (l - 1)) >= sblk;
      }
      if (!valid) sc[ct][r] = -1e30f;
    }
  }
  __syncthreads();  // all waves done reading Ks/Kcs before prob overlay

  // ---- fused multi-level softmax (shared max, level-scaled normalizer) ----
#pragma unroll
  for (int r = 0; r < 8; r++) {
    float mx = -1e30f;
#pragma unroll
    for (int ct = 0; ct < 14; ct++) mx = fmaxf(mx, sc[ct][r]);
    mx = fmaxf(mx, __shfl_xor(mx, 1));
    mx = fmaxf(mx, __shfl_xor(mx, 2));
    mx = fmaxf(mx, __shfl_xor(mx, 4));
    mx = fmaxf(mx, __shfl_xor(mx, 8));
    float den = 0.0f;
#pragma unroll
    for (int ct = 0; ct < 14; ct++) {
      float scale = (ct < 8) ? 1.0f : (float)(8 << (ct - 8));
      float e = __expf(sc[ct][r] - mx);
      sc[ct][r] = e;
      den += scale * e;
    }
    den += __shfl_xor(den, 1);
    den += __shfl_xor(den, 2);
    den += __shfl_xor(den, 4);
    den += __shfl_xor(den, 8);
    float inv = 1.0f / den;
    int i_loc = w * 16 + r + hi * 8;
#pragma unroll
    for (int ct = 0; ct < 14; ct++) {
      float scale = (ct < 8) ? 1.0f : (float)(8 << (ct - 8));
      unsigned short pb = f2bf(scale * sc[ct][r] * inv);
      if (ct < 8)
        P0[i_loc * 128 + ct * 16 + l16] = pb;
      else
        Pc[i_loc * 96 + (ct - 8) * 16 + l16] = pb;
    }
  }
  __syncthreads();

  // ---- stage 2: out = P0 @ V0 (K=128) + Pc @ Vc (K=96) ----
  v8f ot[4];
#pragma unroll
  for (int nt = 0; nt < 4; nt++) ot[nt] = v8f_zero();
  const int ka0 = hi * 16;
#pragma unroll
  for (int ks = 0; ks < 128; ks += 32) {
    Frag af;
    int row = w * 16 + l16;
    af.q[0] = *(const v4u*)&P0[row * 128 + ks + ka0];
    af.q[1] = *(const v4u*)&P0[row * 128 + ks + ka0 + 8];
#pragma unroll
    for (int nt = 0; nt < 4; nt++) {
      Frag bf;
      int d = nt * 16 + l16;
      bf.q[0] = *(const v4u*)&Vst[d * 128 + ks + kb0];
      bf.q[1] = *(const v4u*)&Vst[d * 128 + ks + kb0 + 8];
      ot[nt] = WMMA_BF16(af.v, bf.v, ot[nt]);
    }
  }
#pragma unroll
  for (int ks = 0; ks < 96; ks += 32) {
    Frag af;
    int row = w * 16 + l16;
    af.q[0] = *(const v4u*)&Pc[row * 96 + ks + ka0];
    af.q[1] = *(const v4u*)&Pc[row * 96 + ks + ka0 + 8];
#pragma unroll
    for (int nt = 0; nt < 4; nt++) {
      Frag bf;
      int d = nt * 16 + l16;
      bf.q[0] = *(const v4u*)&Vcst[d * 96 + ks + kb0];
      bf.q[1] = *(const v4u*)&Vcst[d * 96 + ks + kb0 + 8];
      ot[nt] = WMMA_BF16(af.v, bf.v, ot[nt]);
    }
  }
  // ---- write o in [B,T,H*D] (bf16) for the projection GEMM ----
#pragma unroll
  for (int nt = 0; nt < 4; nt++) {
    int d = nt * 16 + l16;
#pragma unroll
    for (int r = 0; r < 8; r++) {
      size_t t = (size_t)g * 64 + w * 16 + r + hi * 8;
      o[((size_t)b * T_SEQ + t) * CE + hOff + d] = f2bf(ot[nt][r]);
    }
  }
}

// ---------------------------------------------------------------------------
extern "C" void kernel_launch(void* const* d_in, const int* in_sizes, int n_in,
                              void* d_out, int out_size, void* d_ws,
                              size_t ws_size, hipStream_t stream) {
  const float* x = (const float*)d_in[0];
  const float* W_attn = (const float*)d_in[1];
  const float* b_attn = (const float*)d_in[2];
  const float* W_proj = (const float*)d_in[3];
  const float* b_proj = (const float*)d_in[4];

  const size_t MR = (size_t)4 * T_SEQ;  // 32768 token rows

  char* ws = (char*)d_ws;
  size_t off = 0;
  auto alloc = [&](size_t bytes) -> void* {
    void* p = ws + off;
    off = (off + bytes + 255) & ~(size_t)255;
    return p;
  };
  unsigned short* x_bf = (unsigned short*)alloc(MR * CE * 2);
  unsigned short* Wa_bf = (unsigned short*)alloc((size_t)CE * 3 * CE * 2);
  unsigned short* Wp_bf = (unsigned short*)alloc((size_t)CE * CE * 2);
  unsigned short* qkv_bf = (unsigned short*)alloc(MR * 3 * CE * 2);
  unsigned short* pk = (unsigned short*)alloc((size_t)64 * 2016 * 64 * 2);
  unsigned short* pv = (unsigned short*)alloc((size_t)64 * 2016 * 64 * 2);
  unsigned short* o_bf = (unsigned short*)alloc(MR * CE * 2);

  {
    int n = (int)(MR * CE);
    cvt_f32_bf16<<<(n + 255) / 256, 256, 0, stream>>>(x, x_bf, n);
  }
  {
    int n = CE * 3 * CE;
    cvt_f32_bf16<<<(n + 255) / 256, 256, 0, stream>>>(W_attn, Wa_bf, n);
  }
  {
    int n = CE * CE;
    cvt_f32_bf16<<<(n + 255) / 256, 256, 0, stream>>>(W_proj, Wp_bf, n);
  }

  // qkv = x @ W_attn + b_attn  -> bf16
  gemm_bf16_wmma<<<dim3(3 * CE / 128, (int)(MR / 128)), 256, 0, stream>>>(
      x_bf, Wa_bf, b_attn, qkv_bf, (int)MR, 3 * CE, CE, 0);

  // pooled K/V summaries (6 levels)
  pool_kv<<<dim3(2016, 64), 64, 0, stream>>>(qkv_bf, pk, pv);

  // fused multipole attention (64KB dynamic LDS)
  fma_attn<<<dim3(128, 64), 128, 65536, stream>>>(qkv_bf, pk, pv, o_bf);

  // out = o @ W_proj + b_proj -> fp32
  gemm_bf16_wmma<<<dim3(CE / 128, (int)(MR / 128)), 256, 0, stream>>>(
      o_bf, Wp_bf, b_proj, d_out, (int)MR, CE, CE, 1);
}
